// MPNN_40896678592682
// MI455X (gfx1250) — compile-verified, matrix operand order checked by reading, not run on hardware
//
#include <hip/hip_runtime.h>
#include <math.h>

// ---------------------------------------------------------------------------
// MPNN (NNConv + GRU) for MI455X / gfx1250, wave32, WMMA f16->f32.
//
// Strategy:
//   * theta = relu(w_e * nn1_w + nn1_b) @ nn2_w + nn2_b  computed ONCE with
//     v_wmma_f32_16x16x32_f16 and stored as f16 (halves HBM traffic for the
//     3 per-step reads: ~1.7 GB total @ 23.3 TB/s ~= 73 us).
//   * per-step message msg_e = out[src_e] @ theta_e is per-edge (unique B per
//     edge -> not WMMA-able): wave-per-edge VALU kernel, scalar-broadcast
//     source features, packed f16x2 theta loads, f32 atomic scatter to dst.
//   * all node GEMMs (lin0, root, GRU gates, lin1, lin2) share one generic
//     WMMA tile kernel (one 16x16 C tile per wave, K steps of 32).
// ---------------------------------------------------------------------------

typedef _Float16 f16;
typedef __attribute__((ext_vector_type(16))) _Float16 v16h;
typedef __attribute__((ext_vector_type(8)))  _Float16 v8h;
typedef __attribute__((ext_vector_type(2)))  _Float16 v2h;
typedef __attribute__((ext_vector_type(8)))  float    v8f;

#define INDIM  128
#define DIM    64
#define EHID   128
#define THN    (DIM * DIM)   // 4096
#define STEPS  3             // reference compile-time constant (can't read device scalar during capture)

// ---------------------------------------------------------------------------
// Generic WMMA GEMM:  C(MxN) = act(A(MxK,f16,row-major) @ Bt(NxK,f16)^T + bias)
// One wave computes one 16x16 C tile. M,N multiples of 16; K multiple of 32.
// Fragment layouts follow CDNA5 ISA 7.12.2 (16-bit A 16x32 / B 32x16 / f32 C).
// ---------------------------------------------------------------------------
__global__ __launch_bounds__(256)
void mpnn_wmma_gemm(const f16* __restrict__ A, const f16* __restrict__ Bt,
                    const float* __restrict__ bias,
                    float* __restrict__ Cf, f16* __restrict__ Ch,
                    int M, int K, int N, int relu)
{
    int wave = (blockIdx.x * blockDim.x + threadIdx.x) >> 5;
    int lane = threadIdx.x & 31;
    int mtiles = M >> 4;
    int ntiles = N >> 4;
    if (wave >= mtiles * ntiles) return;
    int tm = (wave % mtiles) << 4;
    int tn = (wave / mtiles) << 4;

    int r    = lane & 15;   // A row within tile / B column / C column
    int half = lane >> 4;   // K-half selector (ISA A/B 16-bit layout)

    const f16* arow = A  + (size_t)(tm + r) * K + half * 8;
    const f16* brow = Bt + (size_t)(tn + r) * K + half * 8;

    v8f c = {};
    for (int kb = 0; kb < K; kb += 32) {
        v8h alo = *(const v8h*)(arow + kb);        // K = kb + half*8 .. +7
        v8h ahi = *(const v8h*)(arow + kb + 16);   // K = kb + 16 + half*8 .. +7
        v8h blo = *(const v8h*)(brow + kb);
        v8h bhi = *(const v8h*)(brow + kb + 16);
        v16h a, b;
#pragma unroll
        for (int i = 0; i < 8; ++i) {
            a[i] = alo[i]; a[8 + i] = ahi[i];
            b[i] = blo[i]; b[8 + i] = bhi[i];
        }
        c = __builtin_amdgcn_wmma_f32_16x16x32_f16(false, a, false, b,
                                                   (short)0, c, false, false);
    }

    float bv = bias ? bias[tn + r] : 0.0f;
#pragma unroll
    for (int i = 0; i < 8; ++i) {
        int m = tm + i + half * 8;                 // C VGPR i: M=i (lanes 0-15), M=i+8 (16-31)
        float v = c[i] + bv;
        if (relu) v = fmaxf(v, 0.0f);
        size_t off = (size_t)m * N + tn + r;
        if (Cf) Cf[off] = v;
        if (Ch) Ch[off] = (f16)v;
    }
}

// ---------------------------------------------------------------------------
// Small elementwise / conversion kernels
// ---------------------------------------------------------------------------
__global__ void mpnn_transpose_f16(const float* __restrict__ src, f16* __restrict__ dst,
                                   int K, int N)   // src KxN f32 -> dst NxK f16
{
    int idx = blockIdx.x * blockDim.x + threadIdx.x;
    if (idx >= K * N) return;
    int k = idx / N, n = idx % N;
    dst[(size_t)n * K + k] = (f16)src[idx];
}

__global__ void mpnn_to_f16(const float* __restrict__ s, f16* __restrict__ d, int n)
{
    int idx = blockIdx.x * blockDim.x + threadIdx.x;
    if (idx < n) d[idx] = (f16)s[idx];
}

// h_e[k] = relu(w_e * nn1_w[k] + nn1_b[k]), stored f16 for WMMA A operand
__global__ void mpnn_edge_h(const float* __restrict__ ew, const float* __restrict__ w1,
                            const float* __restrict__ b1, f16* __restrict__ H, int E)
{
    int idx = blockIdx.x * blockDim.x + threadIdx.x;
    if (idx >= E * EHID) return;
    int e = idx >> 7, k = idx & 127;
    H[idx] = (f16)fmaxf(ew[e] * w1[k] + b1[k], 0.0f);
}

__global__ void mpnn_deg(const int* __restrict__ ei, float* __restrict__ deg, int E)
{
    int e = blockIdx.x * blockDim.x + threadIdx.x;
    if (e < E) atomicAdd(&deg[ei[E + e]], 1.0f);
}

__global__ void mpnn_invdeg(const float* __restrict__ deg, float* __restrict__ inv, int n)
{
    int i = blockIdx.x * blockDim.x + threadIdx.x;
    if (i < n) inv[i] = deg[i] > 0.0f ? 1.0f / deg[i] : 0.0f;
}

// Wave-per-edge: msg_e = h[src_e] (f32, 64) @ theta_e (f16 64x64), atomic-add
// into agg[dst_e]. Lane l handles output features 2l and 2l+1 (packed f16x2
// theta loads are fully coalesced: 32 lanes cover one 128B theta row).
__global__ __launch_bounds__(256)
void mpnn_edge_msg(const float* __restrict__ h, const f16* __restrict__ theta,
                   const int* __restrict__ ei, float* __restrict__ agg, int E)
{
    int wave = (blockIdx.x * blockDim.x + threadIdx.x) >> 5;
    int lane = threadIdx.x & 31;
    if (wave >= E) return;
    int e   = wave;
    int src = __builtin_amdgcn_readfirstlane(ei[e]);        // uniform -> s_load path
    int dst = __builtin_amdgcn_readfirstlane(ei[E + e]);

    const v2h*   th = (const v2h*)(theta + (size_t)e * THN);  // 32 x f16x2 per row
    const float* hs = h + (size_t)src * DIM;

    float a0 = 0.0f, a1 = 0.0f;
#pragma unroll 4
    for (int d = 0; d < DIM; ++d) {
        float hv = hs[d];                  // wave-uniform scalar broadcast
        v2h   t  = th[d * 32 + lane];
        a0 += hv * (float)t[0];
        a1 += hv * (float)t[1];
    }
    atomicAdd(&agg[(size_t)dst * DIM + lane * 2 + 0], a0);
    atomicAdd(&agg[(size_t)dst * DIM + lane * 2 + 1], a1);
}

// m = relu(agg * inv_deg + (h @ root_w) + conv_b)
__global__ void mpnn_node_m(const float* __restrict__ agg, const float* __restrict__ inv,
                            const float* __restrict__ rootout, const float* __restrict__ cb,
                            float* __restrict__ m, int n)
{
    int idx = blockIdx.x * blockDim.x + threadIdx.x;
    if (idx >= n * DIM) return;
    int v = idx >> 6, f = idx & 63;
    m[idx] = fmaxf(agg[idx] * inv[v] + rootout[idx] + cb[f], 0.0f);
}

// One GRU step (torch gate order r,z,n); in-place update of h.
__global__ void mpnn_gru(const float* __restrict__ gi, const float* __restrict__ gh,
                         float* __restrict__ h, int n)
{
    int idx = blockIdx.x * blockDim.x + threadIdx.x;
    if (idx >= n * DIM) return;
    int v = idx >> 6, f = idx & 63;
    size_t b = (size_t)v * (3 * DIM) + f;
    float r    = 1.0f / (1.0f + expf(-(gi[b]           + gh[b])));
    float z    = 1.0f / (1.0f + expf(-(gi[b + DIM]     + gh[b + DIM])));
    float cand = tanhf(gi[b + 2 * DIM] + r * gh[b + 2 * DIM]);
    h[idx] = (1.0f - z) * cand + z * h[idx];
}

// ---------------------------------------------------------------------------
// Launcher
// ---------------------------------------------------------------------------
static inline size_t alignup(size_t x) { return (x + 255) & ~(size_t)255; }
static inline int cdiv(long long a, long long b) { return (int)((a + b - 1) / b); }

extern "C" void kernel_launch(void* const* d_in, const int* in_sizes, int n_in,
                              void* d_out, int out_size, void* d_ws, size_t ws_size,
                              hipStream_t stream)
{
    const float* x      = (const float*)d_in[0];
    const int*   ei     = (const int*)  d_in[1];
    const float* ew     = (const float*)d_in[2];
    const float* lin0_w = (const float*)d_in[3];
    const float* lin0_b = (const float*)d_in[4];
    const float* nn1_w  = (const float*)d_in[5];
    const float* nn1_b  = (const float*)d_in[6];
    const float* nn2_w  = (const float*)d_in[7];
    const float* nn2_b  = (const float*)d_in[8];
    const float* root_w = (const float*)d_in[9];
    const float* conv_b = (const float*)d_in[10];
    const float* g_wih  = (const float*)d_in[11];
    const float* g_whh  = (const float*)d_in[12];
    const float* g_bih  = (const float*)d_in[13];
    const float* g_bhh  = (const float*)d_in[14];
    const float* lin1_w = (const float*)d_in[15];
    const float* lin1_b = (const float*)d_in[16];
    const float* lin2_w = (const float*)d_in[17];
    const float* lin2_b = (const float*)d_in[18];

    const int N = in_sizes[0] / INDIM;   // 10000
    const int E = in_sizes[2];           // 50000 (edge_weight is Ex1)

    // ---- workspace carve ----
    char* p = (char*)d_ws;
    auto take = [&](size_t bytes) { char* q = p; p += alignup(bytes); return q; };
    f16*   w2t   = (f16*)  take((size_t)THN * EHID * 2);   // nn2_w^T  4096x128
    f16*   lin0t = (f16*)  take((size_t)DIM * INDIM * 2);  // lin0_w^T 64x128
    f16*   roott = (f16*)  take((size_t)DIM * DIM * 2);
    f16*   wiht  = (f16*)  take((size_t)3 * DIM * DIM * 2);
    f16*   whht  = (f16*)  take((size_t)3 * DIM * DIM * 2);
    f16*   lin1t = (f16*)  take((size_t)DIM * DIM * 2);
    f16*   lin2t = (f16*)  take((size_t)DIM * DIM * 2);
    f16*   xh    = (f16*)  take((size_t)N * INDIM * 2);
    f16*   H     = (f16*)  take((size_t)E * EHID * 2);
    f16*   theta = (f16*)  take((size_t)E * THN * 2);      // 410 MB, f16
    float* hbuf  = (float*)take((size_t)N * DIM * 4);
    f16*   hh    = (f16*)  take((size_t)N * DIM * 2);
    float* agg   = (float*)take((size_t)N * DIM * 4);
    float* tmp   = (float*)take((size_t)N * DIM * 4);
    float* mbuf  = (float*)take((size_t)N * DIM * 4);
    f16*   mh    = (f16*)  take((size_t)N * DIM * 2);
    float* gi    = (float*)take((size_t)N * 3 * DIM * 4);
    float* gh    = (float*)take((size_t)N * 3 * DIM * 4);
    float* deg   = (float*)take((size_t)N * 4);
    float* invd  = (float*)take((size_t)N * 4);
    (void)ws_size; (void)n_in; (void)out_size;

    const int TB = 256;
    auto gemm = [&](const f16* A, const f16* Bt, const float* bias,
                    float* Cf, f16* Ch, int M, int K, int Ncols, int relu) {
        long long waves = (long long)(M >> 4) * (Ncols >> 4);
        mpnn_wmma_gemm<<<cdiv(waves * 32, TB), TB, 0, stream>>>(A, Bt, bias, Cf, Ch,
                                                                M, K, Ncols, relu);
    };

    // ---- one-time prep ----
    mpnn_transpose_f16<<<cdiv((long long)EHID * THN, TB), TB, 0, stream>>>(nn2_w, w2t, EHID, THN);
    mpnn_transpose_f16<<<cdiv(INDIM * DIM, TB), TB, 0, stream>>>(lin0_w, lin0t, INDIM, DIM);
    mpnn_transpose_f16<<<cdiv(DIM * DIM, TB), TB, 0, stream>>>(root_w, roott, DIM, DIM);
    mpnn_transpose_f16<<<cdiv(DIM * 3 * DIM, TB), TB, 0, stream>>>(g_wih, wiht, DIM, 3 * DIM);
    mpnn_transpose_f16<<<cdiv(DIM * 3 * DIM, TB), TB, 0, stream>>>(g_whh, whht, DIM, 3 * DIM);
    mpnn_transpose_f16<<<cdiv(DIM * DIM, TB), TB, 0, stream>>>(lin1_w, lin1t, DIM, DIM);
    mpnn_transpose_f16<<<cdiv(DIM * DIM, TB), TB, 0, stream>>>(lin2_w, lin2t, DIM, DIM);
    mpnn_to_f16<<<cdiv((long long)N * INDIM, TB), TB, 0, stream>>>(x, xh, N * INDIM);

    // edge MLP layer 1 (f16) then theta GEMM: E x 128 @ 128 x 4096 (+bias) -> f16
    mpnn_edge_h<<<cdiv((long long)E * EHID, TB), TB, 0, stream>>>(ew, nn1_w, nn1_b, H, E);
    gemm(H, w2t, nn2_b, nullptr, theta, E, EHID, THN, 0);

    // degree / inverse degree
    hipMemsetAsync(deg, 0, (size_t)N * 4, stream);
    mpnn_deg<<<cdiv(E, TB), TB, 0, stream>>>(ei, deg, E);
    mpnn_invdeg<<<cdiv(N, TB), TB, 0, stream>>>(deg, invd, N);

    // out = hid = relu(x @ lin0_w + b)
    gemm(xh, lin0t, lin0_b, hbuf, nullptr, N, INDIM, DIM, 1);
    mpnn_to_f16<<<cdiv((long long)N * DIM, TB), TB, 0, stream>>>(hbuf, hh, N * DIM);

    // ---- message-passing steps ----
    for (int s = 0; s < STEPS; ++s) {
        hipMemsetAsync(agg, 0, (size_t)N * DIM * 4, stream);
        mpnn_edge_msg<<<cdiv((long long)E * 32, TB), TB, 0, stream>>>(hbuf, theta, ei, agg, E);
        gemm(hh, roott, nullptr, tmp, nullptr, N, DIM, DIM, 0);          // h @ root_w
        mpnn_node_m<<<cdiv((long long)N * DIM, TB), TB, 0, stream>>>(agg, invd, tmp, conv_b, mbuf, N);
        mpnn_to_f16<<<cdiv((long long)N * DIM, TB), TB, 0, stream>>>(mbuf, mh, N * DIM);
        gemm(mh, wiht, g_bih, gi, nullptr, N, DIM, 3 * DIM, 0);          // gi = m @ Wih + bih
        gemm(hh, whht, g_bhh, gh, nullptr, N, DIM, 3 * DIM, 0);          // gh = h @ Whh + bhh
        mpnn_gru<<<cdiv((long long)N * DIM, TB), TB, 0, stream>>>(gi, gh, hbuf, N);
        mpnn_to_f16<<<cdiv((long long)N * DIM, TB), TB, 0, stream>>>(hbuf, hh, N * DIM);
    }

    // ---- readout ----
    gemm(hh, lin1t, lin1_b, mbuf, nullptr, N, DIM, DIM, 1);              // relu(h @ lin1 + b)
    mpnn_to_f16<<<cdiv((long long)N * DIM, TB), TB, 0, stream>>>(mbuf, mh, N * DIM);
    gemm(mh, lin2t, lin2_b, (float*)d_out, nullptr, N, DIM, DIM, 0);     // final, f32 out
}
